// Attention_58815282151556
// MI455X (gfx1250) — compile-verified, matrix operand order checked by reading, not run on hardware
//
#include <hip/hip_runtime.h>

#define B_   1024
#define L_   196
#define ATT_ 1024
#define HID_ 1024

typedef __attribute__((ext_vector_type(16))) __bf16 v16bf;
typedef __attribute__((ext_vector_type(8)))  __bf16 v8bf;
typedef __attribute__((ext_vector_type(8)))  float  v8f;

// ---------- fp32 -> bf16 (round to nearest even) ----------
__device__ __forceinline__ unsigned short f2bf(float f) {
    unsigned u = __float_as_uint(f);
    u += 0x7FFFu + ((u >> 16) & 1u);
    return (unsigned short)(u >> 16);
}

__global__ __launch_bounds__(256)
void cvt_bf16_kernel(const float* __restrict__ hprev, const float* __restrict__ wh,
                     unsigned short* __restrict__ hprev_bf, unsigned short* __restrict__ wh_bf) {
    int i = blockIdx.x * blockDim.x + threadIdx.x;
    if (i < B_ * HID_) {
        hprev_bf[i] = f2bf(hprev[i]);
        wh_bf[i]    = f2bf(wh[i]);
    }
}

// ---------- Kernel 1: h_att = h_prev @ W_h^T + b_h via v_wmma_f32_16x16x32_bf16 ----------
// A fragment (16x32 bf16): lane l (r=l&15, kh=l>>4) row r holds
//   elem e   -> K = kh*8 + e ; elem 8+e -> K = 16 + kh*8 + e
// B fragment (32x16 bf16) is symmetric with lane = N column = W_h row (NT GEMM).
// Software-pipelined: fragments for k+32 are fetched while WMMAs for k execute.
__device__ __forceinline__ v16bf load_frag(const __bf16* p /* rowbase + k0 + kh*8 */) {
    v8bf lo = *reinterpret_cast<const v8bf*>(p);
    v8bf hi = *reinterpret_cast<const v8bf*>(p + 16);
    return __builtin_shufflevector(lo, hi, 0,1,2,3,4,5,6,7,8,9,10,11,12,13,14,15);
}

__global__ __launch_bounds__(128)
void hatt_gemm_wmma(const __bf16* __restrict__ A,   // h_prev bf16 [B_, HID_]
                    const __bf16* __restrict__ W,   // W_h    bf16 [ATT_, HID_]
                    const float*  __restrict__ bh,  // [ATT_]
                    float* __restrict__ hatt) {     // [B_, ATT_]
    const int wave  = blockIdx.x * 4 + (threadIdx.x >> 5);  // 0..1023
    const int lane  = threadIdx.x & 31;
    const int r     = lane & 15;
    const int kh    = lane >> 4;
    const int m0    = (wave >> 4) * 16;   // 64 M tiles
    const int n0    = (wave & 15) * 64;   // 16 N strips of 64

    const __bf16* ap  = A + (size_t)(m0 + r) * HID_ + kh * 8;
    const __bf16* bp0 = W + (size_t)(n0 +  0 + r) * HID_ + kh * 8;
    const __bf16* bp1 = W + (size_t)(n0 + 16 + r) * HID_ + kh * 8;
    const __bf16* bp2 = W + (size_t)(n0 + 32 + r) * HID_ + kh * 8;
    const __bf16* bp3 = W + (size_t)(n0 + 48 + r) * HID_ + kh * 8;

    v8f c0 = {}, c1 = {}, c2 = {}, c3 = {};

    // prologue: fragments for k0 = 0
    v16bf a  = load_frag(ap);
    v16bf b0 = load_frag(bp0);
    v16bf b1 = load_frag(bp1);
    v16bf b2 = load_frag(bp2);
    v16bf b3 = load_frag(bp3);

    for (int k0 = 32; k0 < HID_; k0 += 32) {
        // prefetch next k-slice while current WMMAs execute
        v16bf an  = load_frag(ap  + k0);
        v16bf b0n = load_frag(bp0 + k0);
        v16bf b1n = load_frag(bp1 + k0);
        v16bf b2n = load_frag(bp2 + k0);
        v16bf b3n = load_frag(bp3 + k0);
        c0 = __builtin_amdgcn_wmma_f32_16x16x32_bf16(false, a, false, b0, (short)0, c0, false, false);
        c1 = __builtin_amdgcn_wmma_f32_16x16x32_bf16(false, a, false, b1, (short)0, c1, false, false);
        c2 = __builtin_amdgcn_wmma_f32_16x16x32_bf16(false, a, false, b2, (short)0, c2, false, false);
        c3 = __builtin_amdgcn_wmma_f32_16x16x32_bf16(false, a, false, b3, (short)0, c3, false, false);
        a = an; b0 = b0n; b1 = b1n; b2 = b2n; b3 = b3n;
    }
    // epilogue: last k-slice
    c0 = __builtin_amdgcn_wmma_f32_16x16x32_bf16(false, a, false, b0, (short)0, c0, false, false);
    c1 = __builtin_amdgcn_wmma_f32_16x16x32_bf16(false, a, false, b1, (short)0, c1, false, false);
    c2 = __builtin_amdgcn_wmma_f32_16x16x32_bf16(false, a, false, b2, (short)0, c2, false, false);
    c3 = __builtin_amdgcn_wmma_f32_16x16x32_bf16(false, a, false, b3, (short)0, c3, false, false);

    const float bias0 = bh[n0 +  0 + r];
    const float bias1 = bh[n0 + 16 + r];
    const float bias2 = bh[n0 + 32 + r];
    const float bias3 = bh[n0 + 48 + r];
    // C/D layout: VGPR v -> M = m0 + v + 8*kh ; N = r
    #pragma unroll
    for (int v = 0; v < 8; ++v) {
        float* row = hatt + (size_t)(m0 + v + 8 * kh) * ATT_;
        row[n0 +  0 + r] = c0[v] + bias0;
        row[n0 + 16 + r] = c1[v] + bias1;
        row[n0 + 32 + r] = c2[v] + bias2;
        row[n0 + 48 + r] = c3[v] + bias3;
    }
}

// ---------- Kernel 2: scores[b,l] = sum_a relu(h_att[b,a]+fproj[b,l,a])*w_out[a] + b_out ----------
__global__ __launch_bounds__(256)
void scores_kernel(const float* __restrict__ fproj, const float* __restrict__ hatt,
                   const float* __restrict__ wout, const float* __restrict__ bout,
                   float* __restrict__ scores) {
    const int wave = blockIdx.x * 8 + (threadIdx.x >> 5);   // one wave per (b,l)
    const int lane = threadIdx.x & 31;
    const int b = wave / L_;
    const int l = wave % L_;
    const float4* fp = (const float4*)(fproj + ((size_t)b * L_ + l) * ATT_);
    const float4* ha = (const float4*)(hatt + (size_t)b * ATT_);
    const float4* wo = (const float4*)wout;
    float acc = 0.f;
    #pragma unroll
    for (int c = 0; c < 8; ++c) {
        int i = c * 32 + lane;                // 256 float4 per row
        float4 f = fp[i], h = ha[i], w = wo[i];
        acc = fmaf(fmaxf(h.x + f.x, 0.f), w.x, acc);
        acc = fmaf(fmaxf(h.y + f.y, 0.f), w.y, acc);
        acc = fmaf(fmaxf(h.z + f.z, 0.f), w.z, acc);
        acc = fmaf(fmaxf(h.w + f.w, 0.f), w.w, acc);
    }
    #pragma unroll
    for (int off = 16; off > 0; off >>= 1) acc += __shfl_xor(acc, off, 32);
    if (lane == 0) scores[b * L_ + l] = acc + bout[0];
}

// ---------- Kernel 3: softmax over L per batch ----------
__global__ __launch_bounds__(256)
void softmax_kernel(const float* __restrict__ scores, float* __restrict__ alpha) {
    __shared__ float red[256];
    const int b = blockIdx.x, t = threadIdx.x;
    float v = (t < L_) ? scores[b * L_ + t] : -3.4e38f;
    red[t] = v; __syncthreads();
    for (int s = 128; s > 0; s >>= 1) { if (t < s) red[t] = fmaxf(red[t], red[t + s]); __syncthreads(); }
    const float mx = red[0]; __syncthreads();
    float e = (t < L_) ? __expf(v - mx) : 0.f;
    red[t] = e; __syncthreads();
    for (int s = 128; s > 0; s >>= 1) { if (t < s) red[t] += red[t + s]; __syncthreads(); }
    const float inv = 1.f / red[0];
    if (t < L_) alpha[b * L_ + t] = e * inv;
}

// ---------- Kernel 4: context[b,:] = sum_l alpha[b,l] * features[b,l,:] ----------
__global__ __launch_bounds__(256)
void context_kernel(const float* __restrict__ features, const float* __restrict__ alpha,
                    float* __restrict__ out) {
    __shared__ float sa[L_];
    const int b = blockIdx.x, t = threadIdx.x;
    if (t < L_) sa[t] = alpha[b * L_ + t];
    __syncthreads();
    const float4* f = (const float4*)(features + (size_t)b * L_ * ATT_);
    float4 acc = {0.f, 0.f, 0.f, 0.f};
    #pragma unroll 4
    for (int l = 0; l < L_; ++l) {
        const float a = sa[l];
        float4 v = f[(size_t)l * 256 + t];    // 256 float4 per row, thread t owns d = 4t..4t+3
        acc.x = fmaf(a, v.x, acc.x);
        acc.y = fmaf(a, v.y, acc.y);
        acc.z = fmaf(a, v.z, acc.z);
        acc.w = fmaf(a, v.w, acc.w);
    }
    ((float4*)out)[(size_t)b * 256 + t] = acc;
}

extern "C" void kernel_launch(void* const* d_in, const int* in_sizes, int n_in,
                              void* d_out, int out_size, void* d_ws, size_t ws_size,
                              hipStream_t stream) {
    (void)in_sizes; (void)n_in; (void)out_size; (void)ws_size;
    const float* features = (const float*)d_in[0];   // [B,L,ATT]
    const float* fproj    = (const float*)d_in[1];   // [B,L,ATT]
    const float* hprev    = (const float*)d_in[2];   // [B,HID]
    const float* Wh       = (const float*)d_in[3];   // [ATT,HID]
    const float* bh       = (const float*)d_in[4];   // [ATT]
    const float* wout     = (const float*)d_in[5];   // [ATT]
    const float* bout     = (const float*)d_in[6];   // [1]
    float* out            = (float*)d_out;           // [B,ATT]

    char* ws = (char*)d_ws;
    unsigned short* hprev_bf = (unsigned short*)(ws);                                   // 2 MB
    unsigned short* wh_bf    = (unsigned short*)(ws + (size_t)B_ * HID_ * 2);           // 2 MB
    float* hatt   = (float*)(ws + (size_t)4  * 1024 * 1024);                            // 4 MB
    float* scores = (float*)(ws + (size_t)8  * 1024 * 1024);                            // ~0.8 MB
    float* alpha  = (float*)(ws + (size_t)9  * 1024 * 1024);                            // ~0.8 MB

    cvt_bf16_kernel<<<(B_ * HID_ + 255) / 256, 256, 0, stream>>>(hprev, Wh, hprev_bf, wh_bf);
    hatt_gemm_wmma<<<256, 128, 0, stream>>>((const __bf16*)hprev_bf, (const __bf16*)wh_bf, bh, hatt);
    scores_kernel<<<(B_ * L_) / 8, 256, 0, stream>>>(fproj, hatt, wout, bout, scores);
    softmax_kernel<<<B_, 256, 0, stream>>>(scores, alpha);
    context_kernel<<<B_, 256, 0, stream>>>(features, alpha, out);
}
